// LocalGlobalPointAttentionBlock_89086211653967
// MI455X (gfx1250) — compile-verified, matrix operand order checked by reading, not run on hardware
//
#include <hip/hip_runtime.h>
#include <hip/hip_bf16.h>

typedef __attribute__((ext_vector_type(16))) _Float16 v16h;
typedef __attribute__((ext_vector_type(8)))  _Float16 v8h;
typedef __attribute__((ext_vector_type(8)))  float    v8f;

#define NN   4096
#define EE   65536
#define INC  64
#define OUTC 256
#define PEC  32

#if defined(__AMDGCN__) && __has_builtin(__builtin_amdgcn_tensor_load_to_lds) && \
    __has_builtin(__builtin_amdgcn_s_wait_tensorcnt)
#define HAVE_TDM 1
#else
#define HAVE_TDM 0
#endif

// ---------------------------------------------------------------------------
// WMMA helpers
// ---------------------------------------------------------------------------
__device__ __forceinline__ v8f wmma_f16(v16h a, v16h b, v8f c) {
  return __builtin_amdgcn_wmma_f32_16x16x32_f16(false, a, false, b, (short)0, c,
                                                false, false);
}

// Load one 16-lane-row A/B fragment slice for V_WMMA_F32_16X16X32_F16.
// Per ISA layout: lane L (m = L&15, half = L>>4) holds K = [half*8, half*8+8)
// and [half*8+16, half*8+24). p points at row start + k0.
__device__ __forceinline__ v16h load_frag(const _Float16* p, int half) {
  v8h lo = *(const v8h*)(p + half * 8);
  v8h hi = *(const v8h*)(p + half * 8 + 16);
  return __builtin_shufflevector(lo, hi, 0, 1, 2, 3, 4, 5, 6, 7,
                                 8, 9, 10, 11, 12, 13, 14, 15);
}

#if HAVE_TDM
typedef unsigned int u32x4 __attribute__((ext_vector_type(4)));
typedef int          i32x4 __attribute__((ext_vector_type(4)));
typedef int          i32x8 __attribute__((ext_vector_type(8)));

// Issue a 2-D TDM tile load: rows x row_elems f16 elements, global row stride
// row_stride_elems, into LDS at lds_off (tightly packed rows).
// 6-arg builtin form (clang-23 / therock HIP headers): the extra i32x8 is the
// second half of the descriptor SGPR groups (zeros for a <=2-D tensor).
__device__ __forceinline__ void tdm_load_tile_2d(unsigned lds_off,
                                                 const void* gptr,
                                                 unsigned rows,
                                                 unsigned row_elems,
                                                 unsigned row_stride_elems) {
  const unsigned long long ga = (unsigned long long)(uintptr_t)gptr;
  u32x4 g0;
  g0[0] = 1u;                                   // count=1 (valid user D#)
  g0[1] = lds_off;                              // lds_addr
  g0[2] = (unsigned)(ga & 0xffffffffull);       // global_addr[31:0]
  g0[3] = (unsigned)((ga >> 32) & 0x01ffffffull) | 0x80000000u;  // type=2
  i32x8 g1;
  g1[0] = (int)(1u << 16);                      // data_size=1 (2B), mask=0
  g1[1] = (int)((row_elems & 0xffffu) << 16);   // tensor_dim0[15:0]
  g1[2] = (int)((row_elems >> 16) | ((rows & 0xffffu) << 16));  // dim0 hi/dim1 lo
  g1[3] = (int)((rows >> 16) | ((row_elems & 0xffffu) << 16));  // dim1 hi/tile_dim0
  g1[4] = (int)(rows & 0xffffu);                // tile_dim1 (tile_dim2=0)
  g1[5] = (int)row_stride_elems;                // tensor_dim0_stride[31:0]
  g1[6] = 0;
  g1[7] = 0;
  i32x4 z4 = {0, 0, 0, 0};
  i32x8 z8 = {0, 0, 0, 0, 0, 0, 0, 0};
  __builtin_amdgcn_tensor_load_to_lds(g0, g1, z4, z4, z8, 0);
}
#endif

// ---------------------------------------------------------------------------
// f16 WMMA GEMM: out = act(A[M,K] * Bt[N,K]^T + bias) (+resid)
// Each wave computes a 16x64 strip: one A fragment feeds 4 independent
// accumulators (4-way WMMA ILP, 40B/lane/WMMA fragment traffic).
// block = (32,4): 4 waves stacked along M.  grid = (Nc/64, M/64)
// ---------------------------------------------------------------------------
__global__ void gemm_wmma_k(const _Float16* __restrict__ A,
                            const _Float16* __restrict__ Bt,
                            const float* __restrict__ bias,
                            const float* __restrict__ resid,
                            float* __restrict__ outF32,
                            _Float16* __restrict__ outF16,
                            int K, int Nc, int act) {
  const int tn   = blockIdx.x * 64;
  const int tm   = (blockIdx.y * 4 + threadIdx.y) * 16;
  const int lane = threadIdx.x & 31;
  const int half = lane >> 4;
  const int mr   = lane & 15;

  const _Float16* arow = A + (size_t)(tm + mr) * K;
  const _Float16* brow[4];
#pragma unroll
  for (int nt = 0; nt < 4; ++nt)
    brow[nt] = Bt + (size_t)(tn + nt * 16 + mr) * K;

  v8f acc[4];
#pragma unroll
  for (int nt = 0; nt < 4; ++nt) acc[nt] = (v8f){};

  for (int k0 = 0; k0 < K; k0 += 32) {
    const v16h af = load_frag(arow + k0, half);
#pragma unroll
    for (int nt = 0; nt < 4; ++nt) {
      const v16h bf = load_frag(brow[nt] + k0, half);
      acc[nt] = wmma_f16(af, bf, acc[nt]);
    }
  }

#pragma unroll
  for (int nt = 0; nt < 4; ++nt) {
    const int   n  = tn + nt * 16 + mr;
    const float bi = bias ? bias[n] : 0.0f;
#pragma unroll
    for (int r = 0; r < 8; ++r) {
      const int m = tm + half * 8 + r;
      float v = acc[nt][r] + bi;
      if (resid)  v += resid[(size_t)m * Nc + n];
      if (act)    v = fmaxf(v, 0.0f);
      if (outF32) outF32[(size_t)m * Nc + n] = v;
      if (outF16) outF16[(size_t)m * Nc + n] = (_Float16)v;
    }
  }
}

// ---------------------------------------------------------------------------
// Flash attention, H_G=4 heads, HD=64, full N x N softmax.
// grid = (N/64, 4 heads), block = 128 (4 waves). Each wave: 16-row q strip.
// q/k/v/o laid out [N, 256] f16, head h at columns h*64..h*64+63.
// K tile staged to LDS by the Tensor Data Mover (TENSORcnt-tracked);
// V tile staged transposed by the block so PV B-fragments are contiguous.
// ---------------------------------------------------------------------------
__global__ void flash_attn_k(const _Float16* __restrict__ Q,
                             const _Float16* __restrict__ Km,
                             const _Float16* __restrict__ Vm,
                             _Float16* __restrict__ O16, int Nn) {
  const int h    = blockIdx.y;
  const int q0   = blockIdx.x * 64;
  const int wave = threadIdx.x >> 5;
  const int lane = threadIdx.x & 31;
  const int half = lane >> 4;
  const int mr   = lane & 15;
  const int qrow = q0 + wave * 16;
  const size_t ld = OUTC;

  __shared__ _Float16 kT[64 * 64];        // K tile row-major (TDM target)
  __shared__ _Float16 vT[64 * 64];        // V tile transposed: vT[d][key]
  __shared__ _Float16 Pbuf[4][16 * 64];   // per-wave P staging (D->A relayout)

  // Q fragments for this wave's strip, kept in registers across all key tiles
  const _Float16* qr = Q + (size_t)(qrow + mr) * ld + h * 64;
  const v16h qf0 = load_frag(qr + 0, half);
  const v16h qf1 = load_frag(qr + 32, half);

  v8f oacc[4];
  float m_i[8], l_i[8];
#pragma unroll
  for (int i = 0; i < 4; ++i) oacc[i] = (v8f){};
#pragma unroll
  for (int r = 0; r < 8; ++r) { m_i[r] = -3.0e38f; l_i[r] = 0.0f; }

  for (int kt = 0; kt < Nn; kt += 64) {
    __syncthreads();  // previous-tile consumers done before restaging

#if HAVE_TDM
    if (threadIdx.x < 32) {
      tdm_load_tile_2d((unsigned)(uintptr_t)&kT[0],
                       Km + (size_t)kt * ld + h * 64, 64, 64, OUTC);
    }
#endif

    // ---- stage V^T into LDS (cooperative, overlaps the TDM K load) ----
    for (int idx = threadIdx.x; idx < 64 * 8; idx += 128) {
      const int key = idx >> 3;
      const int c8  = idx & 7;
      v8h vv = *(const v8h*)(Vm + (size_t)(kt + key) * ld + h * 64 + c8 * 8);
#pragma unroll
      for (int t = 0; t < 8; ++t) vT[(c8 * 8 + t) * 64 + key] = vv[t];
      if (kt + 64 < Nn)
        __builtin_prefetch(Vm + (size_t)(kt + 64 + key) * ld + h * 64 + c8 * 8,
                           0, 1);
    }

#if HAVE_TDM
    if (threadIdx.x < 32) __builtin_amdgcn_s_wait_tensorcnt(0);
#endif
    __syncthreads();

    // ---- S strip = Q * K^T  (B fragment == rows of K tile) ----
    v8f sacc[4];
#pragma unroll
    for (int nt = 0; nt < 4; ++nt) {
#if HAVE_TDM
      const _Float16* kr = &kT[(nt * 16 + mr) * 64];
#else
      const _Float16* kr = Km + (size_t)(kt + nt * 16 + mr) * ld + h * 64;
#endif
      v16h kf0 = load_frag(kr + 0, half);
      v16h kf1 = load_frag(kr + 32, half);
      v8f s = {};
      s = wmma_f16(qf0, kf0, s);
      s = wmma_f16(qf1, kf1, s);
      sacc[nt] = s;
    }

    // ---- online softmax, per row m = half*8 + r ----
#pragma unroll
    for (int r = 0; r < 8; ++r) {
      float sv[4];
      float mx = -3.0e38f;
#pragma unroll
      for (int nt = 0; nt < 4; ++nt) {
        sv[nt] = sacc[nt][r] * 0.125f;   // 1/sqrt(64)
        mx = fmaxf(mx, sv[nt]);
      }
#pragma unroll
      for (int off = 1; off < 16; off <<= 1)
        mx = fmaxf(mx, __shfl_xor(mx, off));
      const float mnew = fmaxf(m_i[r], mx);
      const float corr = __expf(m_i[r] - mnew);
      const int   m    = half * 8 + r;
      float psum = 0.0f;
#pragma unroll
      for (int nt = 0; nt < 4; ++nt) {
        const float p = __expf(sv[nt] - mnew);
        psum += p;
        Pbuf[wave][m * 64 + nt * 16 + mr] = (_Float16)p;
      }
#pragma unroll
      for (int off = 1; off < 16; off <<= 1)
        psum += __shfl_xor(psum, off);
      l_i[r] = l_i[r] * corr + psum;
      m_i[r] = mnew;
#pragma unroll
      for (int ot = 0; ot < 4; ++ot) oacc[ot][r] *= corr;
    }

    // ---- O += P * V  (A from Pbuf, B from vT rows => contiguous ds loads) --
    const _Float16* prow = &Pbuf[wave][mr * 64];
    const v16h pf0 = load_frag(prow + 0, half);
    const v16h pf1 = load_frag(prow + 32, half);
#pragma unroll
    for (int ot = 0; ot < 4; ++ot) {
      const _Float16* vr = &vT[(ot * 16 + mr) * 64];
      v16h vf0 = load_frag(vr + 0, half);
      v16h vf1 = load_frag(vr + 32, half);
      oacc[ot] = wmma_f16(pf0, vf0, oacc[ot]);
      oacc[ot] = wmma_f16(pf1, vf1, oacc[ot]);
    }
  }

  // ---- normalize + write ----
#pragma unroll
  for (int r = 0; r < 8; ++r) {
    const int   m   = qrow + half * 8 + r;
    const float inv = 1.0f / (l_i[r] + 1e-20f);
#pragma unroll
    for (int ot = 0; ot < 4; ++ot)
      O16[(size_t)m * ld + h * 64 + ot * 16 + mr] = (_Float16)(oacc[ot][r] * inv);
  }
}

// ---------------------------------------------------------------------------
// Edge scatter passes for the PointTransformerConv segment softmax
// ---------------------------------------------------------------------------
__device__ __forceinline__ void atomicMaxF32(float* addr, float v) {
  if (v >= 0.0f) atomicMax((int*)addr, __float_as_int(v));
  else           atomicMin((unsigned int*)addr, __float_as_uint(v));
}

__device__ __forceinline__ float edge_delta(const float* pos, int s, int d,
                                            const float* Wpos, const float* bpos,
                                            int hh, int o) {
  const float px = pos[d * 3 + 0] - pos[s * 3 + 0];
  const float py = pos[d * 3 + 1] - pos[s * 3 + 1];
  const float pz = pos[d * 3 + 2] - pos[s * 3 + 2];
  return px * Wpos[hh * 192 + 0 * 64 + o] + py * Wpos[hh * 192 + 1 * 64 + o] +
         pz * Wpos[hh * 192 + 2 * 64 + o] + bpos[hh * 64 + o];
}

__global__ void edge_pass1_k(const int* __restrict__ ei,
                             const float* __restrict__ pos,
                             const float* __restrict__ a_src,
                             const float* __restrict__ a_dst,
                             const float* __restrict__ Wpos,
                             const float* __restrict__ bpos,
                             float* __restrict__ amax) {
  const size_t idx = (size_t)blockIdx.x * 256 + threadIdx.x;  // E*256 threads
  const int e = (int)(idx >> 8), c = (int)(idx & 255);
  const int s = ei[e], d = ei[EE + e];
  const int hh = c >> 6, o = c & 63;
  const float delta = edge_delta(pos, s, d, Wpos, bpos, hh, o);
  const float alpha =
      a_dst[(size_t)d * 256 + c] - a_src[(size_t)s * 256 + c] + delta;
  atomicMaxF32(amax + (size_t)d * 256 + c, alpha);
}

__global__ void edge_pass2_k(const int* __restrict__ ei,
                             const float* __restrict__ pos,
                             const float* __restrict__ a_src,
                             const float* __restrict__ a_dst,
                             const float* __restrict__ val,
                             const float* __restrict__ Wpos,
                             const float* __restrict__ bpos,
                             const float* __restrict__ amax,
                             float* __restrict__ den,
                             float* __restrict__ num) {
  const size_t idx = (size_t)blockIdx.x * 256 + threadIdx.x;
  const int e = (int)(idx >> 8), c = (int)(idx & 255);
  const int s = ei[e], d = ei[EE + e];
  const int hh = c >> 6, o = c & 63;
  const float delta = edge_delta(pos, s, d, Wpos, bpos, hh, o);
  const float alpha =
      a_dst[(size_t)d * 256 + c] - a_src[(size_t)s * 256 + c] + delta;
  const float ex = __expf(alpha - amax[(size_t)d * 256 + c]);
  atomicAdd(den + (size_t)d * 256 + c, ex);
  atomicAdd(num + (size_t)d * 256 + c, ex * (val[(size_t)s * 256 + c] + delta));
}

__global__ void finalize_xpt_k(const float* __restrict__ num,
                               const float* __restrict__ den,
                               float* __restrict__ out,
                               _Float16* __restrict__ out16) {
  const size_t i = (size_t)blockIdx.x * 256 + threadIdx.x;  // N*256
  const float v = num[i] / (den[i] + 1e-16f);
  out[i]   = v;
  out16[i] = (_Float16)v;
}

// ---------------------------------------------------------------------------
// LayerNorm over 256 channels, one row per 256-thread block
// ---------------------------------------------------------------------------
__global__ void layernorm_k(const float* __restrict__ in,
                            const float* __restrict__ g,
                            const float* __restrict__ b,
                            float* __restrict__ outF32,
                            _Float16* __restrict__ outF16, int outStride) {
  const int row = blockIdx.x, c = threadIdx.x;
  __shared__ float red[256];
  const float x = in[(size_t)row * 256 + c];
  red[c] = x; __syncthreads();
  for (int s = 128; s > 0; s >>= 1) { if (c < s) red[c] += red[c + s]; __syncthreads(); }
  const float mean = red[0] * (1.0f / 256.0f);
  __syncthreads();
  const float dd = x - mean;
  red[c] = dd * dd; __syncthreads();
  for (int s = 128; s > 0; s >>= 1) { if (c < s) red[c] += red[c + s]; __syncthreads(); }
  const float var = red[0] * (1.0f / 256.0f);
  const float y = dd * rsqrtf(var + 1e-5f) * g[c] + b[c];
  if (outF32) outF32[(size_t)row * 256 + c] = y;
  if (outF16) outF16[(size_t)row * outStride + c] = (_Float16)y;
}

// positional-encoding MLP: pos[N,3] -> 32 -> 32, written into hcat[:,256:288]
__global__ void pe_k(const float* __restrict__ pos,
                     const float* __restrict__ W1, const float* __restrict__ b1,
                     const float* __restrict__ W2, const float* __restrict__ b2,
                     _Float16* __restrict__ hcat) {
  const int n = blockIdx.x, j = threadIdx.x;  // 32 threads
  __shared__ float hbuf[32];
  const float p0 = pos[n * 3 + 0], p1 = pos[n * 3 + 1], p2 = pos[n * 3 + 2];
  float hv = p0 * W1[0 * 32 + j] + p1 * W1[1 * 32 + j] + p2 * W1[2 * 32 + j] + b1[j];
  hbuf[j] = fmaxf(hv, 0.0f);
  __syncthreads();
  float o = b2[j];
  for (int t = 0; t < 32; ++t) o += hbuf[t] * W2[t * 32 + j];
  hcat[(size_t)n * 288 + 256 + j] = (_Float16)o;
}

// ---------------------------------------------------------------------------
// Packing / utility kernels
// ---------------------------------------------------------------------------
__global__ void pack_t_k(const float* __restrict__ in, _Float16* __restrict__ out,
                         int K, int Nv) {  // out[n*K+k] = in[k*Nv+n]
  const size_t i = (size_t)blockIdx.x * 256 + threadIdx.x;
  if (i >= (size_t)K * Nv) return;
  const int n = (int)(i / K), k = (int)(i % K);
  out[i] = (_Float16)in[(size_t)k * Nv + n];
}

__global__ void pack_headed_k(const float* __restrict__ in,  // [4,64,64]
                              _Float16* __restrict__ out) {  // [256,64]
  const size_t i = (size_t)blockIdx.x * 256 + threadIdx.x;
  if (i >= 256 * 64) return;
  const int n = (int)(i >> 6), k = (int)(i & 63);
  out[i] = (_Float16)in[(size_t)(n >> 6) * 4096 + k * 64 + (n & 63)];
}

__global__ void cvt_f16_k(const float* __restrict__ in, _Float16* __restrict__ out,
                          size_t n) {
  const size_t i = (size_t)blockIdx.x * 256 + threadIdx.x;
  if (i < n) out[i] = (_Float16)in[i];
}

__global__ void fill_k(float* __restrict__ p, float v, size_t n) {
  const size_t i = (size_t)blockIdx.x * 256 + threadIdx.x;
  if (i < n) p[i] = v;
}

__global__ void copy_f32_k(const float* __restrict__ in, float* __restrict__ out,
                           size_t n) {
  const size_t i = (size_t)blockIdx.x * 256 + threadIdx.x;
  if (i < n) out[i] = in[i];
}

// ---------------------------------------------------------------------------
extern "C" void kernel_launch(void* const* d_in, const int* in_sizes, int n_in,
                              void* d_out, int out_size, void* d_ws, size_t ws_size,
                              hipStream_t stream) {
  (void)in_sizes; (void)n_in; (void)out_size; (void)ws_size;
  const float* x_feat = (const float*)d_in[0];
  const float* pos    = (const float*)d_in[1];
  const int*   ei     = (const int*)d_in[2];
  const float* ptWlin = (const float*)d_in[3];
  const float* ptWsrc = (const float*)d_in[4];
  const float* ptWdst = (const float*)d_in[5];
  const float* ptWpos = (const float*)d_in[6];
  const float* ptbpos = (const float*)d_in[7];
  const float* projW1 = (const float*)d_in[8];   const float* projb1 = (const float*)d_in[9];
  const float* projW2 = (const float*)d_in[10];  const float* projb2 = (const float*)d_in[11];
  const float* lnlg   = (const float*)d_in[12];  const float* lnlb   = (const float*)d_in[13];
  const float* lmW1   = (const float*)d_in[14];  const float* lmb1   = (const float*)d_in[15];
  const float* lmW2   = (const float*)d_in[16];  const float* lmb2   = (const float*)d_in[17];
  const float* n1g    = (const float*)d_in[18];  const float* n1b    = (const float*)d_in[19];
  const float* peW1   = (const float*)d_in[20];  const float* peb1   = (const float*)d_in[21];
  const float* peW2   = (const float*)d_in[22];  const float* peb2   = (const float*)d_in[23];
  const float* combW  = (const float*)d_in[24];  const float* combb  = (const float*)d_in[25];
  const float* qW = (const float*)d_in[26]; const float* qb = (const float*)d_in[27];
  const float* kW = (const float*)d_in[28]; const float* kb = (const float*)d_in[29];
  const float* vW = (const float*)d_in[30]; const float* vb = (const float*)d_in[31];
  const float* oW = (const float*)d_in[32]; const float* ob = (const float*)d_in[33];
  const float* n2g = (const float*)d_in[34]; const float* n2b = (const float*)d_in[35];
  const float* gmW1 = (const float*)d_in[36]; const float* gmb1 = (const float*)d_in[37];
  const float* gmW2 = (const float*)d_in[38]; const float* gmb2 = (const float*)d_in[39];

  // ---- bump allocator over d_ws ----
  char* base = (char*)d_ws;
  size_t off = 0;
  auto alloc = [&](size_t bytes) -> void* {
    void* p = base + off;
    off = (off + bytes + 255) & ~(size_t)255;
    return p;
  };
  const size_t NC = (size_t)NN * OUTC;

  _Float16* Xf16   = (_Float16*)alloc((size_t)NN * INC * 2);
  _Float16* Wlin_t = (_Float16*)alloc(256 * 64 * 2);
  _Float16* Wsrc_t = (_Float16*)alloc(256 * 64 * 2);
  _Float16* Wdst_t = (_Float16*)alloc(256 * 64 * 2);
  _Float16* pW1t = (_Float16*)alloc(256 * 256 * 2);
  _Float16* pW2t = (_Float16*)alloc(256 * 256 * 2);
  _Float16* lW1t = (_Float16*)alloc(256 * 256 * 2);
  _Float16* lW2t = (_Float16*)alloc(256 * 256 * 2);
  _Float16* cWt  = (_Float16*)alloc(256 * 288 * 2);
  _Float16* qWt  = (_Float16*)alloc(256 * 256 * 2);
  _Float16* kWt  = (_Float16*)alloc(256 * 256 * 2);
  _Float16* vWt  = (_Float16*)alloc(256 * 256 * 2);
  _Float16* oWt  = (_Float16*)alloc(256 * 256 * 2);
  _Float16* gW1t = (_Float16*)alloc(256 * 256 * 2);
  _Float16* gW2t = (_Float16*)alloc(256 * 256 * 2);
  float* a_src = (float*)alloc(NC * 4);
  float* a_dst = (float*)alloc(NC * 4);
  float* valb  = (float*)alloc(NC * 4);
  float* amax  = (float*)alloc(NC * 4);
  float* den   = (float*)alloc(NC * 4);
  float* num   = (float*)alloc(NC * 4);
  float*    xpt    = (float*)alloc(NC * 4);
  _Float16* xpt16  = (_Float16*)alloc(NC * 2);
  _Float16* h1_16  = (_Float16*)alloc(NC * 2);
  float*    xpt2   = (float*)alloc(NC * 4);
  float*    xln    = (float*)alloc(NC * 4);
  _Float16* xln16  = (_Float16*)alloc(NC * 2);
  _Float16* lmh16  = (_Float16*)alloc(NC * 2);
  float*    xlocal = (float*)alloc(NC * 4);
  _Float16* hcat   = (_Float16*)alloc((size_t)NN * 288 * 2);
  _Float16* hc16   = (_Float16*)alloc(NC * 2);
  _Float16* q16    = (_Float16*)alloc(NC * 2);
  _Float16* k16    = (_Float16*)alloc(NC * 2);
  _Float16* v16    = (_Float16*)alloc(NC * 2);
  _Float16* o16    = (_Float16*)alloc(NC * 2);
  float*    t_o    = (float*)alloc(NC * 4);
  float*    xg     = (float*)alloc(NC * 4);
  _Float16* xg16   = (_Float16*)alloc(NC * 2);
  _Float16* gmh16  = (_Float16*)alloc(NC * 2);

  const dim3 gB(32, 4);
  const dim3 gG(OUTC / 64, NN / 64);
  auto gemm = [&](const _Float16* A, const _Float16* Bt, const float* bias,
                  const float* resid, float* oF32, _Float16* oF16,
                  int K, int act) {
    gemm_wmma_k<<<gG, gB, 0, stream>>>(A, Bt, bias, resid, oF32, oF16, K, OUTC,
                                       act);
  };
  auto blks = [](size_t n) { return (unsigned)((n + 255) / 256); };

  // ---- stage 0: convert / pack ----
  cvt_f16_k<<<blks((size_t)NN * INC), 256, 0, stream>>>(x_feat, Xf16, (size_t)NN * INC);
  pack_headed_k<<<blks(256 * 64), 256, 0, stream>>>(ptWlin, Wlin_t);
  pack_headed_k<<<blks(256 * 64), 256, 0, stream>>>(ptWsrc, Wsrc_t);
  pack_headed_k<<<blks(256 * 64), 256, 0, stream>>>(ptWdst, Wdst_t);
  pack_t_k<<<blks(256 * 256), 256, 0, stream>>>(projW1, pW1t, 256, 256);
  pack_t_k<<<blks(256 * 256), 256, 0, stream>>>(projW2, pW2t, 256, 256);
  pack_t_k<<<blks(256 * 256), 256, 0, stream>>>(lmW1, lW1t, 256, 256);
  pack_t_k<<<blks(256 * 256), 256, 0, stream>>>(lmW2, lW2t, 256, 256);
  pack_t_k<<<blks(288 * 256), 256, 0, stream>>>(combW, cWt, 288, 256);
  pack_t_k<<<blks(256 * 256), 256, 0, stream>>>(qW, qWt, 256, 256);
  pack_t_k<<<blks(256 * 256), 256, 0, stream>>>(kW, kWt, 256, 256);
  pack_t_k<<<blks(256 * 256), 256, 0, stream>>>(vW, vWt, 256, 256);
  pack_t_k<<<blks(256 * 256), 256, 0, stream>>>(oW, oWt, 256, 256);
  pack_t_k<<<blks(256 * 256), 256, 0, stream>>>(gmW1, gW1t, 256, 256);
  pack_t_k<<<blks(256 * 256), 256, 0, stream>>>(gmW2, gW2t, 256, 256);

  // ---- stage 1: node projections (val / a_src / a_dst) ----
  gemm(Xf16, Wlin_t, nullptr, nullptr, valb,  nullptr, INC, 0);
  gemm(Xf16, Wsrc_t, nullptr, nullptr, a_src, nullptr, INC, 0);
  gemm(Xf16, Wdst_t, nullptr, nullptr, a_dst, nullptr, INC, 0);

  // ---- stage 2: edge segment softmax ----
  fill_k<<<blks(NC), 256, 0, stream>>>(amax, -3.0e38f, NC);
  fill_k<<<blks(NC), 256, 0, stream>>>(den, 0.0f, NC);
  fill_k<<<blks(NC), 256, 0, stream>>>(num, 0.0f, NC);
  edge_pass1_k<<<EE, 256, 0, stream>>>(ei, pos, a_src, a_dst, ptWpos, ptbpos, amax);
  edge_pass2_k<<<EE, 256, 0, stream>>>(ei, pos, a_src, a_dst, valb, ptWpos, ptbpos,
                                       amax, den, num);
  finalize_xpt_k<<<NN, 256, 0, stream>>>(num, den, xpt, xpt16);

  // ---- stage 3: projection MLP ----
  gemm(xpt16, pW1t, projb1, nullptr, nullptr, h1_16, 256, 1);
  gemm(h1_16, pW2t, projb2, nullptr, xpt2, nullptr, 256, 0);

  // ---- stage 4: local LN + local MLP (residual) ----
  layernorm_k<<<NN, 256, 0, stream>>>(xpt2, lnlg, lnlb, xln, xln16, 256);
  gemm(xln16, lW1t, lmb1, nullptr, nullptr, lmh16, 256, 1);
  gemm(lmh16, lW2t, lmb2, xln, xlocal, nullptr, 256, 0);

  // ---- stage 5: n1 LN -> hcat[:, :256], pe -> hcat[:, 256:288] ----
  layernorm_k<<<NN, 256, 0, stream>>>(xlocal, n1g, n1b, nullptr, hcat, 288);
  pe_k<<<NN, 32, 0, stream>>>(pos, peW1, peb1, peW2, peb2, hcat);

  // ---- stage 6: combine + q/k/v ----
  gemm_wmma_k<<<gG, gB, 0, stream>>>(hcat, cWt, combb, nullptr, nullptr, hc16,
                                     288, OUTC, 0);
  gemm(hc16, qWt, qb, nullptr, nullptr, q16, 256, 0);
  gemm(hc16, kWt, kb, nullptr, nullptr, k16, 256, 0);
  gemm(hc16, vWt, vb, nullptr, nullptr, v16, 256, 0);

  // ---- stage 7: global flash attention ----
  flash_attn_k<<<dim3(NN / 64, 4), 128, 0, stream>>>(q16, k16, v16, o16, NN);

  // ---- stage 8: output proj + residual + n2 LN ----
  gemm(o16, oWt, ob, xlocal, t_o, nullptr, 256, 0);
  layernorm_k<<<NN, 256, 0, stream>>>(t_o, n2g, n2b, xg, xg16, 256);

  // ---- stage 9: global MLP (residual) -> d_out ----
  gemm(xg16, gW1t, gmb1, nullptr, nullptr, gmh16, 256, 1);
  gemm(gmh16, gW2t, gmb2, xg, (float*)d_out, nullptr, 256, 0);

  // ---- pos passthrough (second tuple element) ----
  copy_f32_k<<<blks((size_t)NN * 3), 256, 0, stream>>>(
      pos, (float*)d_out + (size_t)NN * OUTC, (size_t)NN * 3);
}
